// Eagle3DraftModel_12360915878790
// MI455X (gfx1250) — matrix operand order repaired and checked by reading, not compiled, not run on hardware
//
#include <hip/hip_runtime.h>
#include <math.h>

// ---------------------------------------------------------------------------
// Eagle3 draft model forward for MI455X (gfx1250, wave32, WMMA).
// GEMM path: fp32 global -> float4 register staging (double-buffered) ->
// packed bf16 LDS tiles (native v_cvt bf16) -> v_wmma_f32_16x16x32_bf16.
// ---------------------------------------------------------------------------

#define DEV __device__ __forceinline__

typedef __attribute__((ext_vector_type(16))) __bf16 v16bf;
typedef __attribute__((ext_vector_type(2)))  __bf16 bf16x2;
typedef __attribute__((ext_vector_type(8)))  float  v8f;

union Frag { unsigned int u[8]; v16bf v; };

static constexpr int BB = 4, SS = 64, PP = 2048, HID = 2048;
static constexpr int NQ = 16, NKV = 4, HD = 128, TT = PP + SS;   // 2112
static constexpr int INTER = 8192, VOCAB = 32000;
static constexpr int M_TOK = BB * SS;                            // 256

static constexpr int LDA = 36;  // ushort row stride: 72B rows, 8B aligned, 18-dword (conflict-free)
static constexpr int LDB = 34;  // ushort row stride: 17-dword (conflict-free reads)

// Native bf16 conversion (RNE) - lets the backend pick packed cvt ops.
DEV unsigned short f2bf(float f) {
  union { __bf16 b; unsigned short u; } cv;
  cv.b = (__bf16)f;
  return cv.u;
}
DEV unsigned int pack2(float lo, float hi) {
  union { bf16x2 v; unsigned int u; } cv;
  cv.v[0] = (__bf16)lo;
  cv.v[1] = (__bf16)hi;
  return cv.u;
}

DEV v8f zero8() {
  v8f z;
#pragma unroll
  for (int i = 0; i < 8; ++i) z[i] = 0.f;
  return z;
}

// A fragment (16x32 bf16): lane m=l&15, h=l>>4; element e -> k=((e&8)<<1)+h*8+(e&7).
template <int LD>
DEV void load_a_frag(const unsigned short* sA, int row_base, int lane, Frag& a) {
  const int m = lane & 15, h = lane >> 4;
  const unsigned short* rowp = sA + (row_base + m) * LD;
#pragma unroll
  for (int p = 0; p < 8; ++p) {
    const int e = 2 * p;
    const int k = ((e & 8) << 1) + h * 8 + (e & 7);
    a.u[p] = *(const unsigned int*)(rowp + k);
  }
}
// B fragment (32x16 bf16): lane n=l&15, h=l>>4; element e -> k=((e&8)<<1)+h*8+(e&7).
// LDS layout: [n][k], row stride LD.
template <int LD>
DEV void load_b_frag(const unsigned short* sB, int col_base, int lane, Frag& b) {
  const int n = col_base + (lane & 15), h = lane >> 4;
  const unsigned short* colp = sB + n * LD + h * 8;
#pragma unroll
  for (int p = 0; p < 8; ++p) b.u[p] = *(const unsigned int*)(colp + (((2 * p) & 8) << 1) + ((2 * p) & 7));
}

DEV v8f wmma_bf16(const Frag& a, const Frag& b, v8f c) {
  return __builtin_amdgcn_wmma_f32_16x16x32_bf16(false, a.v, false, b.v,
                                                 (short)0, c, false, false);
}

// store 8 consecutive bf16 (from 2 float4) at 8B-aligned LDS address
DEV void store_pk8(unsigned short* p, const float4& x, const float4& y) {
  *(uint2*)(p)     = make_uint2(pack2(x.x, x.y), pack2(x.z, x.w));
  *(uint2*)(p + 4) = make_uint2(pack2(y.x, y.y), pack2(y.z, y.w));
}

// ---------------------------------------------------------------------------
// Generic GEMM: C[M,N] = A[M,K] @ B[K,N] (+R), fp32 row-major.
// 256 thr (8 waves as 4x2), block tile 64x128, wave tile 16x64, K-step 32,
// register-staged double buffering.
// ---------------------------------------------------------------------------
__global__ void __launch_bounds__(256)
gemm_rrr(const float* __restrict__ A, const float* __restrict__ B,
         const float* R, float* C, int M, int N, int K) {
  __shared__ __align__(16) unsigned short sA[64 * LDA];
  __shared__ __align__(16) unsigned short sB[128 * LDB];
  const int tid = threadIdx.x, lane = tid & 31, wave = tid >> 5;
  const int wm = wave >> 1, wn = wave & 1;
  const int bm = blockIdx.y * 64, bn = blockIdx.x * 128;

  const int ra = tid >> 2, ca = (tid & 3) * 8;     // A: 64x32/256 = 8 floats
  const int rb = tid >> 5, cb = (tid & 31) * 4;    // B: 32x128/256 = 16 floats

  float4 aR0, aR1;
  float4 bR[4];

  auto load_tiles = [&](int k0) {
    const float* ap = A + (long)(bm + ra) * K + k0 + ca;
    aR0 = *(const float4*)(ap);
    aR1 = *(const float4*)(ap + 4);
#pragma unroll
    for (int j = 0; j < 4; ++j)
      bR[j] = *(const float4*)(B + (long)(k0 + rb + 8 * j) * N + bn + cb);
  };
  auto store_tiles = [&]() {
    store_pk8(&sA[ra * LDA + ca], aR0, aR1);
#pragma unroll
    for (int j = 0; j < 4; ++j) {
      const int r = rb + 8 * j;
      sB[(cb + 0) * LDB + r] = f2bf(bR[j].x);
      sB[(cb + 1) * LDB + r] = f2bf(bR[j].y);
      sB[(cb + 2) * LDB + r] = f2bf(bR[j].z);
      sB[(cb + 3) * LDB + r] = f2bf(bR[j].w);
    }
  };

  v8f acc[4];
#pragma unroll
  for (int t = 0; t < 4; ++t) acc[t] = zero8();

  load_tiles(0);
  store_tiles();
  for (int k0 = 0; k0 < K; k0 += 32) {
    __syncthreads();                       // LDS tile ready
    const bool more = (k0 + 32) < K;
    if (more) load_tiles(k0 + 32);         // next tile loads fly during WMMA
    Frag a; load_a_frag<LDA>(sA, wm * 16, lane, a);
#pragma unroll
    for (int t = 0; t < 4; ++t) {
      Frag b; load_b_frag<LDB>(sB, wn * 64 + t * 16, lane, b);
      acc[t] = wmma_bf16(a, b, acc[t]);
    }
    __syncthreads();                       // all reads done
    if (more) store_tiles();
  }

  const int h = lane >> 4, nn = lane & 15;
#pragma unroll
  for (int t = 0; t < 4; ++t) {
    const int gn = bn + wn * 64 + t * 16 + nn;
#pragma unroll
    for (int r = 0; r < 8; ++r) {
      const int gm = bm + wm * 16 + r + 8 * h;   // D: m=r+8h, n=lane&15
      float v = acc[t][r];
      if (R) v += R[(long)gm * N + gn];
      C[(long)gm * N + gn] = v;
    }
  }
}

// ---------------------------------------------------------------------------
// scores[bh,m,t] = (Q_bh[m,:] . K_bh[t,:]) / sqrt(128)
// 128 thr (4 waves 2x2), block tile 32(m) x 64(t), wave tile 16x32.
// ---------------------------------------------------------------------------
__global__ void __launch_bounds__(128)
attn_scores(const float* __restrict__ q, const float* __restrict__ present,
            float* __restrict__ scores) {
  __shared__ __align__(16) unsigned short sA[32 * LDA];
  __shared__ __align__(16) unsigned short sB[64 * LDA];
  const int tid = threadIdx.x, lane = tid & 31, wave = tid >> 5;
  const int wm = wave >> 1, wn = wave & 1;
  const int bh = blockIdx.z, b = bh >> 2, hkv = bh & 3;
  const int bm = blockIdx.y * 32, bn = blockIdx.x * 64;
  const int gq = bm >> 6, srow = bm & 63;
  const float* kall = present + ((long)((b * 2 + 0) * NKV + hkv)) * TT * HD;
  const int qcol = (hkv * 4 + gq) * HD;

  const int ra = tid >> 2, ca = (tid & 3) * 8;      // A: 32x32/128 = 8 floats
  const int rk = tid >> 1, ck = (tid & 1) * 16;     // B: 64x32/128 = 16 floats
  float4 aR0, aR1, kR0, kR1, kR2, kR3;

  auto load_tiles = [&](int d0) {
    const float* ap = q + (long)(b * SS + srow + ra) * (NQ * HD) + qcol + d0 + ca;
    aR0 = *(const float4*)(ap);
    aR1 = *(const float4*)(ap + 4);
    const float* kp = kall + (long)(bn + rk) * HD + d0 + ck;
    kR0 = *(const float4*)(kp);     kR1 = *(const float4*)(kp + 4);
    kR2 = *(const float4*)(kp + 8); kR3 = *(const float4*)(kp + 12);
  };
  auto store_tiles = [&]() {
    store_pk8(&sA[ra * LDA + ca], aR0, aR1);
    store_pk8(&sB[rk * LDA + ck], kR0, kR1);
    store_pk8(&sB[rk * LDA + ck + 8], kR2, kR3);
  };

  v8f acc[2];
  acc[0] = zero8(); acc[1] = zero8();

  load_tiles(0);
  store_tiles();
  for (int d0 = 0; d0 < HD; d0 += 32) {
    __syncthreads();
    const bool more = (d0 + 32) < HD;
    if (more) load_tiles(d0 + 32);
    Frag a; load_a_frag<LDA>(sA, wm * 16, lane, a);
#pragma unroll
    for (int t = 0; t < 2; ++t) {
      Frag bb; load_b_frag<LDA>(sB, wn * 32 + t * 16, lane, bb);
      acc[t] = wmma_bf16(a, bb, acc[t]);
    }
    __syncthreads();
    if (more) store_tiles();
  }

  const float scale = 0.08838834764831845f;          // 1/sqrt(128)
  const int h = lane >> 4, nn = lane & 15;
#pragma unroll
  for (int t = 0; t < 2; ++t) {
    const int gt = bn + wn * 32 + t * 16 + nn;
#pragma unroll
    for (int r = 0; r < 8; ++r) {
      const int gm = bm + wm * 16 + r + 8 * h;
      scores[((long)bh * 256 + gm) * TT + gt] = acc[t][r] * scale;
    }
  }
}

// ---------------------------------------------------------------------------
// attn = probs(bf16) @ V   (K=2112, N=128).  A staged with zero conversion.
// ---------------------------------------------------------------------------
__global__ void __launch_bounds__(128)
attn_pv(const unsigned short* __restrict__ probs, const float* __restrict__ present,
        float* __restrict__ attn) {
  __shared__ __align__(16) unsigned short sA[32 * LDA];
  __shared__ __align__(16) unsigned short sB[128 * LDB];
  const int tid = threadIdx.x, lane = tid & 31, wave = tid >> 5;
  const int wm = wave >> 1, wn = wave & 1;
  const int bh = blockIdx.z, b = bh >> 2, hkv = bh & 3;
  const int bm = blockIdx.y * 32;
  const float* vall = present + ((long)((b * 2 + 1) * NKV + hkv)) * TT * HD;
  const unsigned short* arow = probs + (long)bh * 256 * TT;

  const int ra = tid >> 2, ca = (tid & 3) * 8;      // A: 32x32/128 = 8 bf16 (16B)
  const int rb = tid >> 5, cb = (tid & 31) * 4;     // B: 32x128/128 = 32 floats
  uint4 aR;
  float4 bR[8];

  auto load_tiles = [&](int k0) {
    aR = *(const uint4*)(arow + (long)(bm + ra) * TT + k0 + ca);
#pragma unroll
    for (int j = 0; j < 8; ++j)
      bR[j] = *(const float4*)(vall + (long)(k0 + rb + 4 * j) * HD + cb);
  };
  auto store_tiles = [&]() {
    unsigned short* p = &sA[ra * LDA + ca];
    *(uint2*)(p)     = make_uint2(aR.x, aR.y);
    *(uint2*)(p + 4) = make_uint2(aR.z, aR.w);
#pragma unroll
    for (int j = 0; j < 8; ++j) {
      const int r = rb + 4 * j;
      sB[(cb + 0) * LDB + r] = f2bf(bR[j].x);
      sB[(cb + 1) * LDB + r] = f2bf(bR[j].y);
      sB[(cb + 2) * LDB + r] = f2bf(bR[j].z);
      sB[(cb + 3) * LDB + r] = f2bf(bR[j].w);
    }
  };

  v8f acc[4];
#pragma unroll
  for (int t = 0; t < 4; ++t) acc[t] = zero8();

  load_tiles(0);
  store_tiles();
  for (int k0 = 0; k0 < TT; k0 += 32) {
    __syncthreads();
    const bool more = (k0 + 32) < TT;
    if (more) load_tiles(k0 + 32);
    Frag a; load_a_frag<LDA>(sA, wm * 16, lane, a);
#pragma unroll
    for (int t = 0; t < 4; ++t) {
      Frag bb; load_b_frag<LDB>(sB, wn * 64 + t * 16, lane, bb);
      acc[t] = wmma_bf16(a, bb, acc[t]);
    }
    __syncthreads();
    if (more) store_tiles();
  }

  const int h = lane >> 4, nn = lane & 15;
#pragma unroll
  for (int t = 0; t < 4; ++t) {
    const int gd = wn * 64 + t * 16 + nn;
#pragma unroll
    for (int r = 0; r < 8; ++r) {
      const int gm = bm + wm * 16 + r + 8 * h;     // gm = gq*64 + s
      attn[(long)(b * SS + (gm & 63)) * (NQ * HD) + (hkv * 4 + (gm >> 6)) * HD + gd]
          = acc[t][r];
    }
  }
}

// ---------------------------------------------------------------------------
// Elementwise / reduction kernels
// ---------------------------------------------------------------------------
__global__ void __launch_bounds__(256)
dual_rms(const float* __restrict__ e, const float* __restrict__ hs,
         const float* __restrict__ w_in, const float* __restrict__ w_hid,
         float* __restrict__ normed) {
  const int row = blockIdx.x, tid = threadIdx.x;
  __shared__ float red[256];
  float s = 0.f;
  for (int c = tid; c < HID; c += 256) { float x = e[(long)row * HID + c]; s += x * x; }
  red[tid] = s; __syncthreads();
  for (int o = 128; o; o >>= 1) { if (tid < o) red[tid] += red[tid + o]; __syncthreads(); }
  float inv = rsqrtf(red[0] / (float)HID + 1e-6f);
  __syncthreads();
  for (int c = tid; c < HID; c += 256)
    normed[(long)row * (2 * HID) + c] = e[(long)row * HID + c] * inv * w_in[c];
  s = 0.f;
  for (int c = tid; c < HID; c += 256) { float x = hs[(long)row * HID + c]; s += x * x; }
  red[tid] = s; __syncthreads();
  for (int o = 128; o; o >>= 1) { if (tid < o) red[tid] += red[tid + o]; __syncthreads(); }
  inv = rsqrtf(red[0] / (float)HID + 1e-6f);
  __syncthreads();
  for (int c = tid; c < HID; c += 256)
    normed[(long)row * (2 * HID) + HID + c] = hs[(long)row * HID + c] * inv * w_hid[c];
}

__global__ void __launch_bounds__(256)
rms_one(const float* __restrict__ in, const float* __restrict__ w,
        float* __restrict__ out) {
  const int row = blockIdx.x, tid = threadIdx.x;
  __shared__ float red[256];
  float s = 0.f;
  for (int c = tid; c < HID; c += 256) { float x = in[(long)row * HID + c]; s += x * x; }
  red[tid] = s; __syncthreads();
  for (int o = 128; o; o >>= 1) { if (tid < o) red[tid] += red[tid + o]; __syncthreads(); }
  const float inv = rsqrtf(red[0] / (float)HID + 1e-6f);
  __syncthreads();
  for (int c = tid; c < HID; c += 256)
    out[(long)row * HID + c] = in[(long)row * HID + c] * inv * w[c];
}

__global__ void __launch_bounds__(256)
rope_kernel(float* __restrict__ buf, const float* __restrict__ rope,
            const int* __restrict__ pos_ids, int nheads) {
  const int idx = blockIdx.x * 256 + threadIdx.x;
  const int total = M_TOK * nheads * 64;
  if (idx >= total) return;
  const int j = idx & 63;
  const int hh = (idx >> 6) % nheads;
  const int row = idx / (64 * nheads);
  const int pos = pos_ids[row];
  const float c = rope[(long)pos * 128 + j];
  const float s = rope[(long)pos * 128 + 64 + j];
  float* p = buf + (long)row * (nheads * HD) + hh * HD;
  const float x1 = p[j], x2 = p[64 + j];
  p[j]      = x1 * c - x2 * s;
  p[64 + j] = x2 * c + x1 * s;
}

__global__ void __launch_bounds__(256)
fill_present(const float* __restrict__ past, const float* __restrict__ kbuf,
             const float* __restrict__ vbuf, float* __restrict__ present) {
  const long idx = (long)blockIdx.x * 256 + threadIdx.x;
  const int d = (int)(idx & 127);
  const int t = (int)((idx >> 7) % TT);
  const int rest = (int)(idx / ((long)128 * TT));
  const int kv = rest & 3, c = (rest >> 2) & 1, b = rest >> 3;
  float val;
  if (t < PP) {
    val = past[((long)((b * 2 + c) * NKV + kv) * PP + t) * HD + d];
  } else {
    const int s = t - PP;
    const float* src = c ? vbuf : kbuf;
    val = src[(long)(b * SS + s) * (NKV * HD) + kv * HD + d];
  }
  present[idx] = val;
}

__global__ void __launch_bounds__(256)
attn_softmax(const float* __restrict__ scores, const int* __restrict__ ctx_len,
             const unsigned char* __restrict__ amask,
             unsigned short* __restrict__ probs) {
  const int row = blockIdx.x;                  // bh*256 + m
  const int bh = row >> 8, m = row & 255;
  const int b = bh >> 2, s = m & 63;
  const float* p = scores + (long)row * TT;
  unsigned short* po = probs + (long)row * TT;
  const int ctx = ctx_len[b];
  const unsigned char* mrow = amask + (long)(b * SS + s) * SS;
  __shared__ float red[256];
  const int tid = threadIdx.x;

  float mx = -3.4e38f;
  for (int t = tid; t < TT; t += 256) {
    const bool valid = (t < PP) ? (t < ctx) : (mrow[t - PP] != 0);
    if (valid) mx = fmaxf(mx, p[t]);
  }
  red[tid] = mx; __syncthreads();
  for (int o = 128; o; o >>= 1) { if (tid < o) red[tid] = fmaxf(red[tid], red[tid + o]); __syncthreads(); }
  mx = red[0]; __syncthreads();

  float sum = 0.f;
  for (int t = tid; t < TT; t += 256) {
    const bool valid = (t < PP) ? (t < ctx) : (mrow[t - PP] != 0);
    if (valid) sum += __expf(p[t] - mx);
  }
  red[tid] = sum; __syncthreads();
  for (int o = 128; o; o >>= 1) { if (tid < o) red[tid] += red[tid + o]; __syncthreads(); }
  const float rinv = 1.f / red[0];
  __syncthreads();

  for (int t = tid; t < TT; t += 256) {
    const bool valid = (t < PP) ? (t < ctx) : (mrow[t - PP] != 0);
    po[t] = f2bf(valid ? __expf(p[t] - mx) * rinv : 0.f);
  }
}

__global__ void __launch_bounds__(256)
silu_mul(const float* __restrict__ g, const float* __restrict__ u,
         float* __restrict__ o, int n) {
  const int idx = blockIdx.x * 256 + threadIdx.x;
  if (idx < n) {
    const float x = g[idx];
    o[idx] = (x / (1.f + __expf(-x))) * u[idx];
  }
}

__global__ void __launch_bounds__(256)
gather_rows(const float* __restrict__ hs, const int* __restrict__ lt,
            float* __restrict__ outg) {
  const int idx = blockIdx.x * 256 + threadIdx.x;   // 256*2048 total
  const int c = idx & (HID - 1);
  const int row = idx >> 11;
  const int b = row >> 6, i = row & 63;
  const int src = lt[b * SS + i];
  outg[idx] = hs[(long)(b * SS + src) * HID + c];
}

__global__ void __launch_bounds__(256)
log_softmax_k(float* __restrict__ logits) {
  const int row = blockIdx.x, tid = threadIdx.x;
  float* p = logits + (long)row * VOCAB;
  __shared__ float red[256];
  float mx = -3.4e38f;
  for (int t = tid; t < VOCAB; t += 256) mx = fmaxf(mx, p[t]);
  red[tid] = mx; __syncthreads();
  for (int o = 128; o; o >>= 1) { if (tid < o) red[tid] = fmaxf(red[tid], red[tid + o]); __syncthreads(); }
  mx = red[0]; __syncthreads();
  float sum = 0.f;
  for (int t = tid; t < VOCAB; t += 256) sum += __expf(p[t] - mx);
  red[tid] = sum; __syncthreads();
  for (int o = 128; o; o >>= 1) { if (tid < o) red[tid] += red[tid + o]; __syncthreads(); }
  const float lse = mx + logf(red[0]);
  __syncthreads();
  for (int t = tid; t < VOCAB; t += 256) p[t] = p[t] - lse;
}

// ---------------------------------------------------------------------------
extern "C" void kernel_launch(void* const* d_in, const int* in_sizes, int n_in,
                              void* d_out, int out_size, void* d_ws, size_t ws_size,
                              hipStream_t stream) {
  (void)in_sizes; (void)n_in; (void)out_size; (void)ws_size;

  const float* embeds = (const float*)d_in[0];
  const float* past   = (const float*)d_in[1];
  const float* rope   = (const float*)d_in[2];
  const int*   ctx    = (const int*)d_in[3];
  const int*   last_ids = (const int*)d_in[5];
  const float* hsi    = (const float*)d_in[6];
  const float* draft  = (const float*)d_in[7];
  const int*   pos_id = (const int*)d_in[8];
  const unsigned char* amask = (const unsigned char*)d_in[9];
  const float* W_fc = (const float*)d_in[10];
  const float* W_q  = (const float*)d_in[11];
  const float* W_k  = (const float*)d_in[12];
  const float* W_v  = (const float*)d_in[13];
  const float* W_o  = (const float*)d_in[14];
  const float* W_gate = (const float*)d_in[15];
  const float* W_up   = (const float*)d_in[16];
  const float* W_down = (const float*)d_in[17];
  const float* W_lm   = (const float*)d_in[18];
  const float* w_hidden_norm = (const float*)d_in[19];
  const float* w_input_ln    = (const float*)d_in[20];
  const float* w_post_ln     = (const float*)d_in[21];
  const float* w_final_norm  = (const float*)d_in[22];

  float* out      = (float*)d_out;
  float* logits   = out;                                   // 4*64*32000
  float* gathered = out + (long)BB * SS * VOCAB;           // 4*64*2048
  float* present  = gathered + (long)BB * SS * HID;        // 4*2*4*2112*128

  float* ws = (float*)d_ws;
  float* hs     = ws; ws += (long)M_TOK * HID;
  float* normed = ws; ws += (long)M_TOK * 2 * HID;
  float* q      = ws; ws += (long)M_TOK * NQ * HD;
  float* k      = ws; ws += (long)M_TOK * NKV * HD;
  float* v      = ws; ws += (long)M_TOK * NKV * HD;
  float* scores = ws; ws += (long)BB * NKV * 256 * TT;
  unsigned short* probs = (unsigned short*)ws;
  ws += ((long)BB * NKV * 256 * TT) / 2;                   // bf16 probs
  float* attn   = ws; ws += (long)M_TOK * NQ * HD;
  float* hn     = ws; ws += (long)M_TOK * HID;
  float* gate   = ws; ws += (long)M_TOK * INTER;
  float* up     = ws; ws += (long)M_TOK * INTER;
  float* act    = ws; ws += (long)M_TOK * INTER;
  float* ng     = ws; ws += (long)M_TOK * HID;

  // hs = draft + hsi @ W_fc
  gemm_rrr<<<dim3(HID / 128, M_TOK / 64), 256, 0, stream>>>(hsi, W_fc, draft, hs,
                                                            M_TOK, HID, 3 * HID);
  // normed = [rms(embeds)*w_in | rms(hs)*w_hid]
  dual_rms<<<M_TOK, 256, 0, stream>>>(embeds, hs, w_input_ln, w_hidden_norm, normed);
  // q/k/v projections
  gemm_rrr<<<dim3(16, 4), 256, 0, stream>>>(normed, W_q, nullptr, q, M_TOK, NQ * HD, 2 * HID);
  gemm_rrr<<<dim3(4, 4), 256, 0, stream>>>(normed, W_k, nullptr, k, M_TOK, NKV * HD, 2 * HID);
  gemm_rrr<<<dim3(4, 4), 256, 0, stream>>>(normed, W_v, nullptr, v, M_TOK, NKV * HD, 2 * HID);
  // rope
  rope_kernel<<<(M_TOK * NQ * 64) / 256, 256, 0, stream>>>(q, rope, pos_id, NQ);
  rope_kernel<<<(M_TOK * NKV * 64) / 256, 256, 0, stream>>>(k, rope, pos_id, NKV);
  // present = concat(past, new k/v)
  fill_present<<<(int)(((long)BB * 2 * NKV * TT * HD) / 256), 256, 0, stream>>>(
      past, k, v, present);
  // attention
  attn_scores<<<dim3(TT / 64, 8, BB * NKV), 128, 0, stream>>>(q, present, scores);
  attn_softmax<<<BB * NKV * 256, 256, 0, stream>>>(scores, ctx, amask, probs);
  attn_pv<<<dim3(1, 8, BB * NKV), 128, 0, stream>>>(probs, present, attn);
  // hs += attn @ W_o
  gemm_rrr<<<dim3(16, 4), 256, 0, stream>>>(attn, W_o, hs, hs, M_TOK, HID, HID);
  // MLP
  rms_one<<<M_TOK, 256, 0, stream>>>(hs, w_post_ln, hn);
  gemm_rrr<<<dim3(INTER / 128, 4), 256, 0, stream>>>(hn, W_gate, nullptr, gate, M_TOK, INTER, HID);
  gemm_rrr<<<dim3(INTER / 128, 4), 256, 0, stream>>>(hn, W_up, nullptr, up, M_TOK, INTER, HID);
  silu_mul<<<(M_TOK * INTER) / 256, 256, 0, stream>>>(gate, up, act, M_TOK * INTER);
  gemm_rrr<<<dim3(16, 4), 256, 0, stream>>>(act, W_down, hs, hs, M_TOK, HID, INTER);
  // head
  gather_rows<<<(M_TOK * HID) / 256, 256, 0, stream>>>(hs, last_ids, gathered);
  rms_one<<<M_TOK, 256, 0, stream>>>(gathered, w_final_norm, ng);
  gemm_rrr<<<dim3(VOCAB / 128, 4), 256, 0, stream>>>(ng, W_lm, nullptr, logits,
                                                     M_TOK, VOCAB, HID);
  log_softmax_k<<<M_TOK, 256, 0, stream>>>(logits);
}
